// LSTM_25254407701009
// MI455X (gfx1250) — matrix operand order repaired and checked by reading, not compiled, not run on hardware
//
#include <hip/hip_runtime.h>

typedef __attribute__((ext_vector_type(16))) _Float16 v16h;
typedef __attribute__((ext_vector_type(8)))  _Float16 v8h;
typedef __attribute__((ext_vector_type(8)))  float    v8f;
typedef _Float16 half_t;

#define T_STEPS 2048
#define BATCH   256
#define HID     512
#define NGROUP  4                 // independent batch groups
#define ROWS    (BATCH / NGROUP)  // 64 batch rows per group
#define NWGH    16                // hidden-split WGs per group
#define UNITS   (HID / NWGH)      // 32 hidden units per WG
#define LCOLS   (4 * UNITS)       // 128 gate columns per WG (i,f,g,o x 32)
#define WSTRIDE (HID + 8)         // LDS W row stride (halves): 260 dwords -> 4 banks apart

// workspace layout:
//   [0, BATCH*HID*2)              : h state, f16, row-major [BATCH][HID]
//   [BATCH*HID*2, +NGROUP*4)      : barrier counters (uint per group)
#define WS_HBYTES (BATCH * HID * 2)

__device__ __forceinline__ float sigmoid_(float x) {
    return 1.0f / (1.0f + __expf(-x));
}
__device__ __forceinline__ float tanh_(float x) {
    return 2.0f / (1.0f + __expf(-2.0f * x)) - 1.0f;   // saturates cleanly
}

// ---- reset the per-group barrier counters (runs every launch) ----
__global__ void bar_init_kernel(unsigned* __restrict__ bar) {
    if (threadIdx.x < NGROUP) bar[threadIdx.x] = 0u;
}

// ---- persistent LSTM: weights LDS-resident, h exchanged via L2 ----
__launch_bounds__(256, 1)
__global__ void lstm_nsplit_kernel(const float* __restrict__ x,      // (T,B)
                                   const float* __restrict__ W_ih,   // (2048,1)
                                   const float* __restrict__ W_hh,   // (2048,512)
                                   const float* __restrict__ b_ih,
                                   const float* __restrict__ b_hh,
                                   const float* __restrict__ W_lin,  // (1,512)
                                   const float* __restrict__ b_lin,
                                   half_t*   __restrict__ hg,        // (256,512) f16 state
                                   unsigned* __restrict__ bar,       // NGROUP counters
                                   float*    __restrict__ out)       // (256,)
{
    __shared__ half_t Wlds[LCOLS * WSTRIDE];   // 128 * 520 * 2B = 130 KB

    const int tid   = threadIdx.x;
    const int wave  = tid >> 5;
    const int lane  = tid & 31;
    const int lhalf = lane >> 4;
    const int l16   = lane & 15;
    const int grp   = blockIdx.x >> 4;   // batch group 0..3
    const int w     = blockIdx.x & 15;   // hidden-split member 0..15
    const int mt    = wave >> 1;         // M-tile 0..3
    const int hs    = wave & 1;          // 16-unit sub-block 0..1
    const int row0  = grp * ROWS + mt * 16;

    // one-time: convert this WG's W_hh slice to f16 in LDS
    // local col lc = g*32 + u  <->  global gate col g*512 + w*32 + u
    for (int idx = tid; idx < LCOLS * HID; idx += blockDim.x) {
        int lc = idx >> 9;            // idx / 512
        int k  = idx & (HID - 1);
        int g  = lc >> 5;
        int u  = lc & 31;
        int gcol = g * HID + w * UNITS + u;
        Wlds[lc * WSTRIDE + k] = (half_t)W_hh[(size_t)gcol * HID + k];
    }

    // per-lane per-gate constants for this wave's 4 N-tiles
    float wih[4], bias[4];
    const half_t* bfrag[4];
#pragma unroll
    for (int g = 0; g < 4; ++g) {
        int gcol = g * HID + w * UNITS + hs * 16 + l16;
        wih[g]  = W_ih[gcol];
        bias[g] = b_ih[gcol] + b_hh[gcol];
        int lc  = g * 32 + hs * 16 + l16;
        bfrag[g] = &Wlds[lc * WSTRIDE + lhalf * 8];
    }

    // A fragment base in the global h state (row = row0 + l16)
    const half_t* afrag = hg + (size_t)(row0 + l16) * HID + lhalf * 8;

    v8f cc;
#pragma unroll
    for (int v = 0; v < 8; ++v) cc[v] = 0.0f;

    __syncthreads();   // LDS weights ready

    for (int t = 0; t < T_STEPS; ++t) {
        // ---- x projection folded into accumulator init ----
        float xv = x[t * BATCH + row0 + l16];
        float xr[8];
#pragma unroll
        for (int v = 0; v < 8; ++v)
            xr[v] = __shfl(xv, v + lhalf * 8, 32);

        v8f acc[4];
#pragma unroll
        for (int g = 0; g < 4; ++g)
#pragma unroll
            for (int v = 0; v < 8; ++v)
                acc[g][v] = fmaf(wih[g], xr[v], bias[g]);

        // ---- gates += h @ W_hh.T : A from L2 (h), B from LDS (weights) ----
        if (t > 0) {
#pragma unroll 4
            for (int k0 = 0; k0 < HID; k0 += 32) {
                v8h alo = *(const v8h*)(afrag + k0);        // K=k0+lhalf*8..+8
                v8h ahi = *(const v8h*)(afrag + k0 + 16);   // K=k0+16+lhalf*8..+8
                v16h a;
#pragma unroll
                for (int j = 0; j < 8; ++j) { a[j] = alo[j]; a[8 + j] = ahi[j]; }

#pragma unroll
                for (int g = 0; g < 4; ++g) {
                    v8h blo = *(const v8h*)(bfrag[g] + k0);
                    v8h bhi = *(const v8h*)(bfrag[g] + k0 + 16);
                    v16h b;
#pragma unroll
                    for (int j = 0; j < 8; ++j) { b[j] = blo[j]; b[8 + j] = bhi[j]; }
                    acc[g] = __builtin_amdgcn_wmma_f32_16x16x32_f16(
                        false, a, false, b, (short)0, acc[g], false, false);
                }
            }
        }

        // ---- wave-local LSTM cell update (i,f,g,o lane-aligned) ----
        float hn[8];
#pragma unroll
        for (int v = 0; v < 8; ++v) {
            float ig = sigmoid_(acc[0][v]);
            float fg = sigmoid_(acc[1][v]);
            float gg = tanh_(acc[2][v]);
            float og = sigmoid_(acc[3][v]);
            float cn = fmaf(fg, cc[v], ig * gg);
            cc[v] = cn;
            hn[v] = og * tanh_(cn);
        }

        // ---- publish this WG's h slice ----
        int colb = w * UNITS + hs * 16 + l16;
#pragma unroll
        for (int v = 0; v < 8; ++v)
            hg[(size_t)(row0 + v + lhalf * 8) * HID + colb] = (half_t)hn[v];

        // ---- per-group barrier (16 WGs), release/acquire fenced ----
        __threadfence();            // stores visible at device scope
        __syncthreads();
        if (tid == 0) {
            atomicAdd(&bar[grp], 1u);
            unsigned target = (unsigned)(NWGH * (t + 1));
            while (__hip_atomic_load(&bar[grp], __ATOMIC_RELAXED,
                                     __HIP_MEMORY_SCOPE_AGENT) < target)
                __builtin_amdgcn_s_sleep(2);
        }
        __syncthreads();
        __threadfence();            // invalidate stale h lines in WGP$
    }

    // ---- final linear: one WG per group writes its 64 outputs ----
    if (w == 0 && tid < ROWS) {
        int row = grp * ROWS + tid;
        float s = b_lin[0];
        for (int k = 0; k < HID; ++k)
            s = fmaf((float)hg[(size_t)row * HID + k], W_lin[k], s);
        out[row] = s;
    }
}

extern "C" void kernel_launch(void* const* d_in, const int* in_sizes, int n_in,
                              void* d_out, int out_size, void* d_ws, size_t ws_size,
                              hipStream_t stream) {
    const float* x     = (const float*)d_in[0];
    const float* W_ih  = (const float*)d_in[1];
    const float* W_hh  = (const float*)d_in[2];
    const float* b_ih  = (const float*)d_in[3];
    const float* b_hh  = (const float*)d_in[4];
    const float* W_lin = (const float*)d_in[5];
    const float* b_lin = (const float*)d_in[6];
    float* out = (float*)d_out;

    half_t*   hg  = (half_t*)d_ws;
    unsigned* bar = (unsigned*)((char*)d_ws + WS_HBYTES);

    bar_init_kernel<<<1, 64, 0, stream>>>(bar);
    lstm_nsplit_kernel<<<NGROUP * NWGH, 256, 0, stream>>>(
        x, W_ih, W_hh, b_ih, b_hh, W_lin, b_lin, hg, bar, out);
}
